// MolNet_Layer_84207128806060
// MI455X (gfx1250) — matrix operand order repaired and checked by reading, not compiled, not run on hardware
//
#include <hip/hip_runtime.h>
#include <math.h>

typedef __bf16 bf16_t;
typedef __attribute__((ext_vector_type(16))) __bf16 v16bf;
typedef __attribute__((ext_vector_type(8)))  __bf16 v8bf;
typedef __attribute__((ext_vector_type(8)))  float  v8f;

#define WPB 4  // waves per block (wave32)

union FragU { v16bf v; v8bf h[2]; };

// ---------------- helpers ----------------

__device__ __forceinline__ bf16_t f2bf(float f) { return (bf16_t)f; }

__device__ __forceinline__ float siluf(float x) {
  return x / (1.0f + __expf(-x));
}

__device__ __forceinline__ float tanh_fast(float x) {
  // 1 - 2/(e^{2x}+1): saturates to +/-1, ~4 VALU ops
  float e = __expf(2.0f * x);
  return 1.0f - 2.0f / (e + 1.0f);
}

// ISA 16-bit A layout (16x32 fragment, K-chunk kc of a [16 x 128] tile):
//   lane<16  holds k_local {0..7, 16..23} + kc*32
//   lane>=16 holds k_local {8..15, 24..31} + kc*32
// Runs of 8 are 8-aligned and never straddle the k=64 concat boundary.

// A-fragment from pre-converted bf16 node tables (no VALU conversion).
__device__ __forceinline__ v16bf load_cat_frag_bf(const bf16_t* bcol, const bf16_t* brow,
                                                  int kc, int lane) {
  const int a = kc * 32 + ((lane & 16) ? 8 : 0);
  FragU u;
#pragma unroll
  for (int run = 0; run < 2; ++run) {
    const int k = a + run * 16;
    const bf16_t* src = (k < 64) ? (bcol + k) : (brow + (k - 64));
    u.h[run] = *(const v8bf*)src;          // one 16B load per run
  }
  return u.v;
}

// Fallback: gather f32 and convert (compiler-chosen bf16 cvt).
__device__ __forceinline__ v16bf load_cat_frag_f32(const float* bcol, const float* brow,
                                                   int kc, int lane) {
  const int a = kc * 32 + ((lane & 16) ? 8 : 0);
  v16bf out;
#pragma unroll
  for (int run = 0; run < 2; ++run) {
    const int k = a + run * 16;
    const float* src = (k < 64) ? (bcol + k) : (brow + (k - 64));
    const float4 q0 = *(const float4*)(src + 0);
    const float4 q1 = *(const float4*)(src + 4);
    const int j = run * 8;
    out[j + 0] = f2bf(q0.x); out[j + 1] = f2bf(q0.y);
    out[j + 2] = f2bf(q0.z); out[j + 3] = f2bf(q0.w);
    out[j + 4] = f2bf(q1.x); out[j + 5] = f2bf(q1.y);
    out[j + 6] = f2bf(q1.z); out[j + 7] = f2bf(q1.w);
  }
  return out;
}

// A-fragment from the per-wave LDS intermediate [16 x 128] bf16 tile.
__device__ __forceinline__ v16bf load_inter_frag(const bf16_t* it, int kc, int lane) {
  const int m = lane & 15;
  const int a = kc * 32 + ((lane & 16) ? 8 : 0);
  FragU u;
  u.h[0] = *(const v8bf*)(it + m * 128 + a);
  u.h[1] = *(const v8bf*)(it + m * 128 + a + 16);
  return u.v;
}

// [16x128] @ [128x64] via 16 WMMAs; B fragments pre-packed per-lane in wpk.
__device__ __forceinline__ void mm_acc(v8f acc[4], const v16bf a[4],
                                       const bf16_t* __restrict__ wpk, int w, int lane) {
#pragma unroll
  for (int kc = 0; kc < 4; ++kc) {
#pragma unroll
    for (int nt = 0; nt < 4; ++nt) {
      const v16bf b = *(const v16bf*)(wpk + ((((w * 4 + kc) * 4 + nt) * 32 + lane) << 4));
      acc[nt] = __builtin_amdgcn_wmma_f32_16x16x32_bf16(
          false, a[kc], false, b, (short)0, acc[nt], false, false);
    }
  }
}

// ---------------- kernels ----------------

__global__ void zero_kernel(float* p, int n) {
  int i = blockIdx.x * blockDim.x + threadIdx.x;
  if (i < n) p[i] = 0.0f;
}

__global__ void deg_kernel(const int* __restrict__ eidx, const float* __restrict__ eattr,
                           float* __restrict__ deg, int E_) {
  int e = blockIdx.x * blockDim.x + threadIdx.x;
  if (e < E_) atomicAdd(&deg[eidx[E_ + e]], eattr[e]);
}

__global__ void dis_kernel(float* deg, int N_) {
  int n = blockIdx.x * blockDim.x + threadIdx.x;
  if (n < N_) { float d = deg[n]; deg[n] = (d > 0.0f) ? rsqrtf(d) : 0.0f; }
}

// Pre-convert node feature tables to bf16 (once per launch, per-node).
__global__ void convert_nodes_kernel(const float* __restrict__ scalar,
                                     const float* __restrict__ vec,
                                     bf16_t* __restrict__ sbf, bf16_t* __restrict__ vbf,
                                     int N_) {
  int i = blockIdx.x * blockDim.x + threadIdx.x;
  const int ns = N_ * 64;
  const int nv = N_ * 192;
  if (i < ns) sbf[i] = f2bf(scalar[i]);
  else if (i < ns + nv) vbf[i - ns] = f2bf(vec[i - ns]);
}

// Pack W (row-major [64 out x 128 in], used as B[k][n] = W[n*128+k]) into the
// B-fragment lane layout: frag (w,kc,nt), lane<16: k {0..7,16..23}, lane>=16: k {8..15,24..31}.
__global__ void pack_weights_kernel(const float* __restrict__ W0, const float* __restrict__ W1,
                                    const float* __restrict__ W2, const float* __restrict__ W3,
                                    const float* __restrict__ W4, const float* __restrict__ W5,
                                    bf16_t* __restrict__ wpk) {
  int t = blockIdx.x * blockDim.x + threadIdx.x;
  if (t >= 6 * 4 * 4 * 32) return;
  const int lane = t & 31;
  const int nt = (t >> 5) & 3;
  const int kc = (t >> 7) & 3;
  const int w  = t >> 9;
  const float* W = (w == 0) ? W0 : (w == 1) ? W1 : (w == 2) ? W2
                 : (w == 3) ? W3 : (w == 4) ? W4 : W5;
  const int n = nt * 16 + (lane & 15);
  const int kbase = kc * 32 + ((lane & 16) ? 8 : 0);
  bf16_t* dst = wpk + (size_t)t * 16;
#pragma unroll
  for (int j = 0; j < 16; ++j) dst[j] = f2bf(W[n * 128 + kbase + (j < 8 ? j : j + 8)]);
}

template <bool TAB>
__global__ void __launch_bounds__(WPB * 32)
edge_kernel(const float* __restrict__ scalar, const float* __restrict__ vec,
            const bf16_t* __restrict__ sbf, const bf16_t* __restrict__ vbf,
            const float* __restrict__ pos, const int* __restrict__ eidx,
            const float* __restrict__ eattr, const float* __restrict__ dis,
            const bf16_t* __restrict__ wpk,
            const float* __restrict__ bss, const float* __restrict__ bvs,
            const float* __restrict__ bsl, const float* __restrict__ bsv,
            const float* __restrict__ bvv, const float* __restrict__ bvl,
            float* __restrict__ s_agg, float* __restrict__ v_agg, int E_) {
  __shared__ bf16_t sh_inter[WPB][16 * 128];   // per-wave intermediate tile
  __shared__ float  sh_pos[WPB][3][16];
  __shared__ float  sh_nrm[WPB][16];
  __shared__ int    sh_col[WPB][16];

  const int wave = threadIdx.x >> 5;
  const int lane = threadIdx.x & 31;
  const int tile = blockIdx.x * WPB + wave;
  if (tile * 16 >= E_) return;                 // wave-uniform exit

  const int m  = lane & 15;
  const int hi = lane >> 4;
  const int nl = lane & 15;
  const int e = tile * 16 + m;
  const bool valid = (e < E_);
  const int ec = valid ? e : (E_ - 1);
  const int rn = eidx[ec];                     // source (row)
  const int cn = eidx[E_ + ec];                // target (col)
  const float nrm = valid ? (dis[rn] * dis[cn] * eattr[ec]) : 0.0f;

  if (hi == 0) {
    sh_nrm[wave][m] = nrm;
    sh_col[wave][m] = cn;
#pragma unroll
    for (int d = 0; d < 3; ++d)
      sh_pos[wave][d][m] = pos[cn * 3 + d] - pos[rn * 3 + d];
  }
  asm volatile("s_wait_dscnt 0" ::: "memory");

  float nrm8[8], pos8[3][8];
  int cn8[8];
#pragma unroll
  for (int r = 0; r < 8; ++r) {
    nrm8[r] = sh_nrm[wave][hi * 8 + r];
    cn8[r]  = sh_col[wave][hi * 8 + r];
#pragma unroll
    for (int d = 0; d < 3; ++d) pos8[d][r] = sh_pos[wave][d][hi * 8 + r];
  }

  float bss_n[4], bvs_n[4], bsl_n[4], bsv_n[4], bvv_n[4], bvl_n[4];
#pragma unroll
  for (int nt = 0; nt < 4; ++nt) {
    bss_n[nt] = bss[nt * 16 + nl];  bvs_n[nt] = bvs[nt * 16 + nl];
    bsl_n[nt] = bsl[nt * 16 + nl];  bsv_n[nt] = bsv[nt * 16 + nl];
    bvv_n[nt] = bvv[nt * 16 + nl];  bvl_n[nt] = bvl[nt * 16 + nl];
  }

  // s_cat fragments: kept resident, reused by s2s and s2v
  v16bf afs[4];
#pragma unroll
  for (int kc = 0; kc < 4; ++kc) {
    if constexpr (TAB)
      afs[kc] = load_cat_frag_bf(sbf + (size_t)cn * 64, sbf + (size_t)rn * 64, kc, lane);
    else
      afs[kc] = load_cat_frag_f32(scalar + (size_t)cn * 64, scalar + (size_t)rn * 64, kc, lane);
  }

  const v8f vz = {0.f, 0.f, 0.f, 0.f, 0.f, 0.f, 0.f, 0.f};
  bf16_t* it = sh_inter[wave];

  // ---------------- scalar path ----------------
  v8f acc_s2s[4] = {vz, vz, vz, vz};
  mm_acc(acc_s2s, afs, wpk, 0, lane);          // s_cat @ Wss^T

  v8f acc_v2s[4] = {vz, vz, vz, vz};
#pragma unroll
  for (int d = 0; d < 3; ++d) {
    v16bf afv[4];
#pragma unroll
    for (int kc = 0; kc < 4; ++kc) {
      if constexpr (TAB)
        afv[kc] = load_cat_frag_bf(vbf + (size_t)cn * 192 + d * 64,
                                   vbf + (size_t)rn * 192 + d * 64, kc, lane);
      else
        afv[kc] = load_cat_frag_f32(vec + (size_t)cn * 192 + d * 64,
                                    vec + (size_t)rn * 192 + d * 64, kc, lane);
    }
    v8f t[4] = {vz, vz, vz, vz};
    mm_acc(t, afv, wpk, 1, lane);              // v_cat @ Wvs^T
#pragma unroll
    for (int nt = 0; nt < 4; ++nt)
#pragma unroll
      for (int r = 0; r < 8; ++r)
        acc_v2s[nt][r] += (t[nt][r] + bvs_n[nt]) * pos8[d][r];
  }

  // activate + pack concat([s2s, v2s]) into LDS tile
#pragma unroll
  for (int nt = 0; nt < 4; ++nt) {
    const int n = nt * 16 + nl;
#pragma unroll
    for (int r = 0; r < 8; ++r) {
      const int mm = hi * 8 + r;
      it[mm * 128 + n]      = f2bf(siluf(acc_s2s[nt][r] + bss_n[nt]));
      it[mm * 128 + 64 + n] = f2bf(siluf(acc_v2s[nt][r]));
    }
  }
  asm volatile("s_wait_dscnt 0" ::: "memory");

  {
    v16bf ai[4];
#pragma unroll
    for (int kc = 0; kc < 4; ++kc) ai[kc] = load_inter_frag(it, kc, lane);
    v8f am[4] = {vz, vz, vz, vz};
    mm_acc(am, ai, wpk, 2, lane);              // concat @ Wsl^T
#pragma unroll
    for (int nt = 0; nt < 4; ++nt) {
      const int n = nt * 16 + nl;
#pragma unroll
      for (int r = 0; r < 8; ++r) {
        const float v = (am[nt][r] + bsl_n[nt]) * nrm8[r];
        atomicAdd(&s_agg[(size_t)cn8[r] * 64 + n], v);
      }
    }
  }

  // ---------------- vector path ----------------
  v8f acc_s2v[4] = {vz, vz, vz, vz};
  mm_acc(acc_s2v, afs, wpk, 3, lane);          // s_cat @ Wsv^T

#pragma unroll
  for (int d = 0; d < 3; ++d) {
    v16bf afv[4];
#pragma unroll
    for (int kc = 0; kc < 4; ++kc) {
      if constexpr (TAB)
        afv[kc] = load_cat_frag_bf(vbf + (size_t)cn * 192 + d * 64,
                                   vbf + (size_t)rn * 192 + d * 64, kc, lane);
      else
        afv[kc] = load_cat_frag_f32(vec + (size_t)cn * 192 + d * 64,
                                    vec + (size_t)rn * 192 + d * 64, kc, lane);
    }
    v8f av[4] = {vz, vz, vz, vz};
    mm_acc(av, afv, wpk, 4, lane);             // v_cat @ Wvv^T

    // pack concat([v2v, tanh(s2v * pos_d)])
#pragma unroll
    for (int nt = 0; nt < 4; ++nt) {
      const int n = nt * 16 + nl;
#pragma unroll
      for (int r = 0; r < 8; ++r) {
        const int mm = hi * 8 + r;
        it[mm * 128 + n]      = f2bf(av[nt][r] + bvv_n[nt]);
        it[mm * 128 + 64 + n] = f2bf(tanh_fast((acc_s2v[nt][r] + bsv_n[nt]) * pos8[d][r]));
      }
    }
    asm volatile("s_wait_dscnt 0" ::: "memory");

    v16bf ai[4];
#pragma unroll
    for (int kc = 0; kc < 4; ++kc) ai[kc] = load_inter_frag(it, kc, lane);
    v8f am[4] = {vz, vz, vz, vz};
    mm_acc(am, ai, wpk, 5, lane);              // concat @ Wvl^T
#pragma unroll
    for (int nt = 0; nt < 4; ++nt) {
      const int n = nt * 16 + nl;
#pragma unroll
      for (int r = 0; r < 8; ++r) {
        const float v = (am[nt][r] + bvl_n[nt]) * nrm8[r];
        atomicAdd(&v_agg[(size_t)cn8[r] * 192 + d * 64 + n], v);
      }
    }
  }
}

__global__ void finalize_kernel(const float* __restrict__ scalar,
                                const float* __restrict__ vec,
                                float* __restrict__ out, int N_) {
  int i = blockIdx.x * blockDim.x + threadIdx.x;
  const int ns = N_ * 64;
  const int nv = N_ * 192;
  if (i < ns) {
    out[i] = siluf(out[i]) + scalar[i];
  } else if (i < ns + nv) {
    out[i] = tanh_fast(out[i]) + vec[i - ns];
  }
}

// ---------------- launch ----------------

static inline size_t align256(size_t x) { return (x + 255) & ~(size_t)255; }

extern "C" void kernel_launch(void* const* d_in, const int* in_sizes, int n_in,
                              void* d_out, int out_size, void* d_ws, size_t ws_size,
                              hipStream_t stream) {
  const float* scalar = (const float*)d_in[0];
  const float* vec    = (const float*)d_in[1];
  const float* pos    = (const float*)d_in[2];
  const int*   eidx   = (const int*)d_in[3];
  const float* eattr  = (const float*)d_in[4];
  const float* Wss = (const float*)d_in[5];  const float* bss = (const float*)d_in[6];
  const float* Wvs = (const float*)d_in[7];  const float* bvs = (const float*)d_in[8];
  const float* Wsl = (const float*)d_in[9];  const float* bsl = (const float*)d_in[10];
  const float* Wsv = (const float*)d_in[11]; const float* bsv = (const float*)d_in[12];
  const float* Wvv = (const float*)d_in[13]; const float* bvv = (const float*)d_in[14];
  const float* Wvl = (const float*)d_in[15]; const float* bvl = (const float*)d_in[16];

  const int N_ = in_sizes[0] / 64;   // scalar is [N, 64]
  const int E_ = in_sizes[4];        // edge_attr is [E]

  const size_t off_deg = 0;
  const size_t off_wpk = align256((size_t)N_ * 4);
  const size_t off_sbf = align256(off_wpk + 6 * 4 * 4 * 32 * 16 * sizeof(bf16_t));
  const size_t off_vbf = align256(off_sbf + (size_t)N_ * 64 * sizeof(bf16_t));
  const size_t need    = off_vbf + (size_t)N_ * 192 * sizeof(bf16_t);
  const bool   tab     = (ws_size >= need);

  float*  deg = (float*)((char*)d_ws + off_deg);
  bf16_t* wpk = (bf16_t*)((char*)d_ws + off_wpk);
  bf16_t* sbf = (bf16_t*)((char*)d_ws + off_sbf);
  bf16_t* vbf = (bf16_t*)((char*)d_ws + off_vbf);

  float* outf  = (float*)d_out;
  float* s_agg = outf;                          // [N,64]
  float* v_agg = outf + (size_t)N_ * 64;        // [N,3,64]

  const int tot = N_ * 256;
  zero_kernel<<<(tot + 255) / 256, 256, 0, stream>>>(outf, tot);
  zero_kernel<<<(N_ + 255) / 256, 256, 0, stream>>>(deg, N_);

  pack_weights_kernel<<<(6 * 4 * 4 * 32 + 127) / 128, 128, 0, stream>>>(
      Wss, Wvs, Wsl, Wsv, Wvv, Wvl, wpk);
  deg_kernel<<<(E_ + 255) / 256, 256, 0, stream>>>(eidx, eattr, deg, E_);
  dis_kernel<<<(N_ + 255) / 256, 256, 0, stream>>>(deg, N_);
  if (tab) {
    convert_nodes_kernel<<<(tot + 255) / 256, 256, 0, stream>>>(scalar, vec, sbf, vbf, N_);
  }

  const int tiles  = (E_ + 15) / 16;
  const int blocks = (tiles + WPB - 1) / WPB;
  if (tab) {
    edge_kernel<true><<<blocks, WPB * 32, 0, stream>>>(
        scalar, vec, sbf, vbf, pos, eidx, eattr, deg, wpk,
        bss, bvs, bsl, bsv, bvv, bvl, s_agg, v_agg, E_);
  } else {
    edge_kernel<false><<<blocks, WPB * 32, 0, stream>>>(
        scalar, vec, sbf, vbf, pos, eidx, eattr, deg, wpk,
        bss, bvs, bsl, bsv, bvv, bvl, s_agg, v_agg, E_);
  }

  finalize_kernel<<<(tot + 255) / 256, 256, 0, stream>>>(scalar, vec, outf, N_);
}
